// SelfAttention_79989470921173
// MI455X (gfx1250) — compile-verified, matrix operand order checked by reading, not run on hardware
//
#include <hip/hip_runtime.h>

typedef __attribute__((ext_vector_type(16))) _Float16 v16h;
typedef __attribute__((ext_vector_type(8)))  _Float16 v8h;
typedef __attribute__((ext_vector_type(8)))  float    v8f;

#define DIM   1024
#define HEADS 16
#define HD    64
#define NSEQ  2048
#define NB    2
#define EPSLN 1e-5f

// ---------- WMMA fragment helpers (CDNA5 16x16x32 f16 layouts) ----------
// A (16x32 f16): lane half picks klo = (lane<16 ? 0 : 8); vector elements
// 0..7 -> k = klo..klo+7, elements 8..15 -> k = klo+16..klo+23.
// B (32x16 f16): lane%16 = column, same k striping.
// C/D (16x16 f32, 8 VGPRs): element r -> row = r + 8*(lane>=16), col = lane%16.

__device__ __forceinline__ v16h frag_cat(v8h lo, v8h hi) {
  v16h r;
#pragma unroll
  for (int i = 0; i < 8; i++) { r[i] = lo[i]; r[i + 8] = hi[i]; }
  return r;
}

__device__ __forceinline__ v16h load_frag(const _Float16* p, int klo) {
  v8h lo = *(const v8h*)(p + klo);        // 16B contiguous
  v8h hi = *(const v8h*)(p + klo + 16);   // 16B contiguous
  return frag_cat(lo, hi);
}

__device__ __forceinline__ v8f wmma_f16(v16h a, v16h b, v8f c) {
  // (neg_a, A, neg_b, B, c_mod, C, reuse_a, reuse_b)
  return __builtin_amdgcn_wmma_f32_16x16x32_f16(false, a, false, b, (short)0, c,
                                                false, false);
}

__device__ __forceinline__ float wave_sum32(float v) {
#pragma unroll
  for (int m = 16; m >= 1; m >>= 1) v += __shfl_xor(v, m, 32);
  return v;
}

// ---------- dtype conversion ----------
__global__ void cvt_f16_kernel(const float* __restrict__ in,
                               _Float16* __restrict__ out, size_t n) {
  size_t i = (size_t)blockIdx.x * blockDim.x + threadIdx.x;
  if (i < n) out[i] = (_Float16)in[i];
}

// in[R][C] f32 -> out[C][R] f16 (weights pre-transposed so WMMA B-fragments
// become contiguous row reads)
__global__ void transpose_cvt_kernel(const float* __restrict__ in,
                                     _Float16* __restrict__ out, int R, int C) {
  size_t i = (size_t)blockIdx.x * blockDim.x + threadIdx.x;
  if (i < (size_t)R * C) {
    int r = (int)(i / C), c = (int)(i % C);
    out[(size_t)c * R + r] = (_Float16)in[i];
  }
}

// ---------- WMMA GEMM: C[M,N] = A[M,K] * BT[N,K]^T (+bias) ----------
// block = 256 thr = 8 waves; block tile 64x128; wave tile 16x64.
template <bool OUT_F16, bool HAS_BIAS>
__global__ void gemm_wmma_kernel(const _Float16* __restrict__ A,
                                 const _Float16* __restrict__ BT,
                                 void* __restrict__ Cout,
                                 const float* __restrict__ bias,
                                 int M, int N, int K) {
  const int lane = threadIdx.x & 31;
  const int w    = threadIdx.x >> 5;
  const int lid  = lane & 15;
  const int half = lane >> 4;
  const int klo  = half * 8;
  const int rowBase = blockIdx.y * 64 + (w & 3) * 16;
  const int colBase = blockIdx.x * 128 + (w >> 2) * 64;

  const _Float16* arow = A + (size_t)(rowBase + lid) * K;
  const _Float16* bcol[4];
#pragma unroll
  for (int t = 0; t < 4; t++) bcol[t] = BT + (size_t)(colBase + t * 16 + lid) * K;

  v8f acc[4] = {};
  for (int k0 = 0; k0 < K; k0 += 32) {
    __builtin_prefetch(arow + k0 + 512, 0, 0);   // global_prefetch_b8
    v16h a = load_frag(arow + k0, klo);
#pragma unroll
    for (int t = 0; t < 4; t++)
      acc[t] = wmma_f16(a, load_frag(bcol[t] + k0, klo), acc[t]);
  }

#pragma unroll
  for (int r = 0; r < 8; r++) {
    const int row = rowBase + r + 8 * half;
#pragma unroll
    for (int t = 0; t < 4; t++) {
      const int col = colBase + t * 16 + lid;
      float v = acc[t][r];
      if (HAS_BIAS) v += bias[col];
      if (OUT_F16)
        ((_Float16*)Cout)[(size_t)row * N + col] = (_Float16)v;
      else
        ((float*)Cout)[(size_t)row * N + col] = v;
    }
  }
}

// ---------- per-head LayerNorm + RoPE ----------
// Emits q,k f16 in [B,H,N,HD]; v pre-TRANSPOSED to [B,H,HD,N] so attention
// can load V^T A-fragments straight from global with contiguous key runs.
__global__ void ln_rope_kernel(const _Float16* __restrict__ qkvh,
                               const float* __restrict__ freq,
                               const float* __restrict__ gq,
                               const float* __restrict__ bq,
                               const float* __restrict__ gk,
                               const float* __restrict__ bk,
                               _Float16* __restrict__ qh,
                               _Float16* __restrict__ kh,
                               _Float16* __restrict__ vT) {
  const int lane = threadIdx.x & 31;
  const int wv   = threadIdx.x >> 5;
  const int hidx = blockIdx.x * 8 + wv;       // (b*N + n)*HEADS + h
  const int h = hidx & (HEADS - 1);
  const int t = hidx >> 4;                    // token index b*N + n
  const int n = t & (NSEQ - 1);
  const int b = t >> 11;                      // NSEQ = 2^11
  const int d0 = lane * 2, d1 = d0 + 1;       // RoPE pair owned by this lane
  const int bh = b * HEADS + h;

  const _Float16* base = qkvh + (size_t)t * (3 * DIM) + h * HD;
  float q0 = (float)base[d0],           q1 = (float)base[d1];
  float k0 = (float)base[DIM + d0],     k1 = (float)base[DIM + d1];
  float v0 = (float)base[2 * DIM + d0], v1 = (float)base[2 * DIM + d1];

  const float cs = freq[((size_t)n * 32 + lane) * 2 + 0];
  const float sn = freq[((size_t)n * 32 + lane) * 2 + 1];

  // LayerNorm(q) over head dim (64 values across 32 lanes, 2/lane)
  float mu = wave_sum32(q0 + q1) * (1.0f / HD);
  float a0 = q0 - mu, a1 = q1 - mu;
  float var = wave_sum32(a0 * a0 + a1 * a1) * (1.0f / HD);
  float rs = rsqrtf(var + EPSLN);
  float nq0 = a0 * rs * gq[d0] + bq[d0];
  float nq1 = a1 * rs * gq[d1] + bq[d1];
  float oq0 = nq0 * cs - nq1 * sn;
  float oq1 = nq0 * sn + nq1 * cs;

  // LayerNorm(k)
  mu = wave_sum32(k0 + k1) * (1.0f / HD);
  a0 = k0 - mu; a1 = k1 - mu;
  var = wave_sum32(a0 * a0 + a1 * a1) * (1.0f / HD);
  rs = rsqrtf(var + EPSLN);
  float nk0 = a0 * rs * gk[d0] + bk[d0];
  float nk1 = a1 * rs * gk[d1] + bk[d1];
  float ok0 = nk0 * cs - nk1 * sn;
  float ok1 = nk0 * sn + nk1 * cs;

  const size_t o = (((size_t)bh) * NSEQ + n) * HD;
  qh[o + d0] = (_Float16)oq0; qh[o + d1] = (_Float16)oq1;
  kh[o + d0] = (_Float16)ok0; kh[o + d1] = (_Float16)ok1;
  // transposed V: [bh][hd][n]
  vT[((size_t)bh * HD + d0) * NSEQ + n] = (_Float16)v0;
  vT[((size_t)bh * HD + d1) * NSEQ + n] = (_Float16)v1;
}

// ---------- flash attention (transposed): one wave per 16-row Q block ----------
// Computes S^T = K * Q^T and O^T = V^T * P^T.  The C-layout of S^T puts one
// query per lane, so softmax stats are per-lane scalars (one shfl_xor(16) to
// merge halves), and P^T re-enters the WMMA as a B-fragment purely in
// registers (the C-layout key striping equals the B-fragment k striping).
// No LDS, no barriers.
__global__ void attn_kernel(const _Float16* __restrict__ qh,
                            const _Float16* __restrict__ kh,
                            const _Float16* __restrict__ vT,
                            _Float16* __restrict__ Oh) {
  const int lane = threadIdx.x & 31;
  const int wv   = threadIdx.x >> 5;
  const int lid  = lane & 15, half = lane >> 4, klo = half * 8;
  const int wid  = blockIdx.x * 8 + wv;
  const int qblk = wid & ((NSEQ / 16) - 1);   // 128 Q blocks
  const int bh   = wid >> 7;                  // 0..31
  const int b    = bh >> 4, h = bh & 15;

  const _Float16* kb = kh + (size_t)bh * NSEQ * HD;
  const _Float16* vb = vT + (size_t)bh * HD * NSEQ;

  // Q as B-fragments (column q = lid), loaded once
  const _Float16* qrow = qh + ((size_t)bh * NSEQ + qblk * 16 + lid) * HD;
  const v16h bQ0 = load_frag(qrow, klo);        // hd 0..31
  const v16h bQ1 = load_frag(qrow + 32, klo);   // hd 32..63

  v8f o[4] = {};                 // O^T tiles: row hd = t*16 + r + 8*half, col q = lid
  float m = -3.0e38f, l = 0.0f;  // per-lane (per-query) softmax state

  for (int kc = 0; kc < NSEQ; kc += 32) {
    // S^T for 32 keys: A = K rows, B = Q rows.  st[s] elem r -> key kc+s*16+r+8*half
    const _Float16* kp0 = kb + (size_t)(kc + lid) * HD;
    const _Float16* kp1 = kb + (size_t)(kc + 16 + lid) * HD;
    __builtin_prefetch(kp0 + 32 * HD, 0, 0);   // next chunk
    v8f s0 = {}, s1 = {};
    s0 = wmma_f16(load_frag(kp0, klo), bQ0, s0);
    s0 = wmma_f16(load_frag(kp0 + 32, klo), bQ1, s0);
    s1 = wmma_f16(load_frag(kp1, klo), bQ0, s1);
    s1 = wmma_f16(load_frag(kp1 + 32, klo), bQ1, s1);

    // per-lane softmax over this lane's 16 keys, then merge with lane^16
    float x0[8], x1[8], mloc = -3.0e38f;
#pragma unroll
    for (int r = 0; r < 8; r++) {
      x0[r] = s0[r] * 0.125f;                  // 1/sqrt(64)
      x1[r] = s1[r] * 0.125f;
      mloc = fmaxf(mloc, fmaxf(x0[r], x1[r]));
    }
    mloc = fmaxf(mloc, __shfl_xor(mloc, 16, 32));
    const float mn = fmaxf(m, mloc);
    const float alpha = __expf(m - mn);
    float sloc = 0.0f;
    v16h bP;                                   // P^T as B-fragment, in registers
#pragma unroll
    for (int r = 0; r < 8; r++) {
      float p0 = __expf(x0[r] - mn);           // key klo + r
      float p1 = __expf(x1[r] - mn);           // key klo + 16 + r
      sloc += p0 + p1;
      bP[r] = (_Float16)p0;
      bP[r + 8] = (_Float16)p1;
    }
    sloc += __shfl_xor(sloc, 16, 32);
    l = l * alpha + sloc;
    m = mn;

    // O^T = V^T * P^T ; A = V^T rows (hd), contiguous keys from global
#pragma unroll
    for (int t = 0; t < 4; t++) {
      const _Float16* vp = vb + (size_t)(t * 16 + lid) * NSEQ + kc;
      v16h aV = load_frag(vp, klo);
#pragma unroll
      for (int r = 0; r < 8; r++) o[t][r] *= alpha;
      o[t] = wmma_f16(aV, bP, o[t]);
    }
  }

  // normalize and write: lane owns query row n = qblk*16 + lid; per tile the
  // 8 accumulator elements are contiguous hd -> one 16B store each
  const float inv = 1.0f / l;
  _Float16* orow = Oh + ((size_t)b * NSEQ + qblk * 16 + lid) * DIM + h * HD;
#pragma unroll
  for (int t = 0; t < 4; t++) {
    v8h ov;
#pragma unroll
    for (int r = 0; r < 8; r++) ov[r] = (_Float16)(o[t][r] * inv);
    *(v8h*)(orow + t * 16 + 8 * half) = ov;
  }
}

// ---------- launcher ----------
extern "C" void kernel_launch(void* const* d_in, const int* in_sizes, int n_in,
                              void* d_out, int out_size, void* d_ws,
                              size_t ws_size, hipStream_t stream) {
  (void)in_sizes; (void)n_in; (void)out_size; (void)ws_size;
  const float* x     = (const float*)d_in[0];
  const float* freq  = (const float*)d_in[1];
  const float* Wqkv  = (const float*)d_in[2];
  const float* gq    = (const float*)d_in[3];
  const float* bq    = (const float*)d_in[4];
  const float* gk    = (const float*)d_in[5];
  const float* bk    = (const float*)d_in[6];
  const float* Wproj = (const float*)d_in[7];
  const float* bproj = (const float*)d_in[8];

  char* w = (char*)d_ws;
  _Float16* xh     = (_Float16*)w; w += (size_t)NB * NSEQ * DIM * 2;
  _Float16* WqkvT  = (_Float16*)w; w += (size_t)3 * DIM * DIM * 2;
  _Float16* WprojT = (_Float16*)w; w += (size_t)DIM * DIM * 2;
  _Float16* qkvh   = (_Float16*)w; w += (size_t)NB * NSEQ * 3 * DIM * 2;
  _Float16* qh     = (_Float16*)w; w += (size_t)NB * NSEQ * DIM * 2;
  _Float16* kh     = (_Float16*)w; w += (size_t)NB * NSEQ * DIM * 2;
  _Float16* vT     = (_Float16*)w; w += (size_t)NB * NSEQ * DIM * 2;
  _Float16* Oh     = (_Float16*)w; w += (size_t)NB * NSEQ * DIM * 2;

  const int M = NB * NSEQ;  // 4096 tokens

  cvt_f16_kernel<<<(M * DIM) / 256, 256, 0, stream>>>(x, xh, (size_t)M * DIM);
  transpose_cvt_kernel<<<(DIM * 3 * DIM) / 256, 256, 0, stream>>>(Wqkv, WqkvT,
                                                                  DIM, 3 * DIM);
  transpose_cvt_kernel<<<(DIM * DIM) / 256, 256, 0, stream>>>(Wproj, WprojT,
                                                              DIM, DIM);

  // QKV projection: [4096,1024] x [1024,3072] -> f16
  dim3 g1(3 * DIM / 128, M / 64);
  gemm_wmma_kernel<true, false><<<g1, 256, 0, stream>>>(xh, WqkvT, qkvh,
                                                        nullptr, M, 3 * DIM, DIM);

  // per-head LN + RoPE; q,k -> [B,H,N,HD], v -> [B,H,HD,N]
  ln_rope_kernel<<<(M * HEADS) / 8, 256, 0, stream>>>(qkvh, freq, gq, bq, gk,
                                                      bk, qh, kh, vT);

  // flash attention: 4096 independent waves, 8 waves per 256-thread block
  attn_kernel<<<(NB * HEADS * (NSEQ / 16)) / 8, 256, 0, stream>>>(qh, kh, vT, Oh);

  // output projection + bias -> f32 d_out
  dim3 g2(DIM / 128, M / 64);
  gemm_wmma_kernel<false, true><<<g2, 256, 0, stream>>>(Oh, WprojT, d_out,
                                                        bproj, M, DIM, DIM);
}
